// BiLSTMCRF_86182813761976
// MI455X (gfx1250) — matrix operand order repaired and checked by reading, not compiled, not run on hardware
//
#include <hip/hip_runtime.h>
#include <math.h>

typedef __attribute__((ext_vector_type(16))) _Float16 v16h;
typedef __attribute__((ext_vector_type(8)))  float    v8f;
typedef __attribute__((ext_vector_type(4)))  unsigned int v4u;
typedef __attribute__((ext_vector_type(4)))  int      v4i;
typedef __attribute__((ext_vector_type(8)))  int      v8i;

#define NB   8      // batch
#define LT   512    // time steps
#define ED   1024   // embed dim
#define HD   1024   // hidden dim
#define KT_  16     // tags
#define G4   4096   // 4*HD
#define ROWS 4096   // NB*LT

#if __has_builtin(__builtin_amdgcn_sched_barrier)
#define SCHED_FENCE() __builtin_amdgcn_sched_barrier(0)
#else
#define SCHED_FENCE()
#endif

#define WMMA_F16(A, B, C) \
  __builtin_amdgcn_wmma_f32_16x16x32_f16(false, (A), false, (B), (short)0, (C), false, false)

// ---------------- workspace layout (bytes) ----------------
static const size_t OFF_AX   = 0;                                   // X as WMMA A-tiles: 256 mt x 32 kt x 512 halves
static const size_t SZ_AX    = (size_t)256*32*512*2;                // 8 MB
static const size_t OFF_BIH  = OFF_AX + SZ_AX;                      // W_ih f/b as B-tiles: 2 x 32kt x 256nt x 512 halves
static const size_t SZ_BIH   = (size_t)2*32*256*512*2;              // 16 MB
static const size_t OFF_BHH  = OFF_BIH + SZ_BIH;                    // W_hh f/b as B-tiles
static const size_t SZ_BHH   = SZ_BIH;                              // 16 MB
static const size_t OFF_G    = OFF_BHH + SZ_BHH;                    // gates_x f16 [2][4096][4096]
static const size_t SZ_G     = (size_t)2*ROWS*G4*2;                 // 64 MB
static const size_t OFF_HS   = OFF_G + SZ_G;                        // hf/hb f16 [2][8][512][1024]
static const size_t SZ_HS    = (size_t)2*NB*LT*HD*2;                // 16 MB
static const size_t OFF_HSWZ = OFF_HS + SZ_HS;                      // h state, A-swizzled, double buffered [2buf][2dir][32kt][512]
static const size_t SZ_HSWZ  = (size_t)2*2*32*512*2;                // 128 KB
static const size_t OFF_EMIT = OFF_HSWZ + SZ_HSWZ;                  // emit f32 [8][512][16]
static const size_t SZ_EMIT  = (size_t)NB*LT*KT_*4;                 // 256 KB
static const size_t OFF_CTR  = OFF_EMIT + SZ_EMIT;                  // grid barrier counter
static const size_t SZ_CTR   = 256;

__device__ __forceinline__ float sigm(float x) { return 1.0f / (1.0f + expf(-x)); }

// ---------------------------------------------------------------------------
// TDM: load 32 rows of 1KB (stride 256KB) from global into LDS at lds_off.
// 2D descriptor, data_size=8B: tile_dim0=128, tile_dim1=32, dim0_stride=32768.
// ---------------------------------------------------------------------------
__device__ __forceinline__ void tdm_load_whh_slab(const void* gsrc, unsigned lds_off) {
#if __has_builtin(__builtin_amdgcn_tensor_load_to_lds)
  unsigned long long ga = (unsigned long long)(uintptr_t)gsrc;
  v4u g0;
  g0[0] = 1u;                                            // count=1 valid user D#
  g0[1] = lds_off;                                       // lds_addr (bytes)
  g0[2] = (unsigned)(ga & 0xFFFFFFFFu);                  // global_addr[31:0]
  g0[3] = (unsigned)((ga >> 32) & 0x01FFFFFFu) | (2u << 30);  // addr[56:32] | type=2
  v8i g1;
  g1[0] = (int)(3u << 16);         // data_size = 8 bytes
  g1[1] = (int)(128u << 16);       // tensor_dim0 = 128 units (bits 79:48 lo)
  g1[2] = (int)(8192u << 16);      // tensor_dim1 = 8192 rows (bits 111:80)
  g1[3] = (int)(128u << 16);       // tile_dim0 = 128 units (bits 127:112)
  g1[4] = 32;                      // tile_dim1 = 32 rows
  g1[5] = 32768;                   // tensor_dim0_stride = 32768 units (256KB)
  g1[6] = 0;
  g1[7] = 0;
  v4i z4 = {0, 0, 0, 0};
#if defined(__clang_major__) && (__clang_major__ >= 23)
  v8i z8 = {0, 0, 0, 0, 0, 0, 0, 0};
  __builtin_amdgcn_tensor_load_to_lds(g0, g1, z4, z4, z8, 0);
#else
  __builtin_amdgcn_tensor_load_to_lds(g0, g1, z4, z4, 0);
#endif
#endif
}

// ---------------------------------------------------------------------------
// Swizzle X (f32 [4096][1024]) into WMMA 16x32 f16 A-tiles.
// ---------------------------------------------------------------------------
__global__ void __launch_bounds__(256) k_swz_A(const float* __restrict__ X,
                                               _Float16* __restrict__ out) {
  int tid  = blockIdx.x * blockDim.x + threadIdx.x;   // 262144 threads
  int lane = tid & 31;
  int tile = tid >> 5;                                // tile = mt*32 + kt
  int mt = tile >> 5, kt = tile & 31;
  int m  = lane & 15;
  int hi = lane >> 4;
  const float* src = X + (size_t)(mt*16 + m) * ED + kt*32;
  _Float16* dst = out + (size_t)tile*512 + lane*16;
#pragma unroll
  for (int i = 0; i < 16; ++i) {
    int k = (i < 8) ? (hi*8 + i) : (16 + hi*8 + (i - 8));
    dst[i] = (_Float16)src[k];
  }
}

// ---------------------------------------------------------------------------
// Swizzle W (f32 [4096][1024] row-major [n][k]) into WMMA 32x16 f16 B-tiles.
// ---------------------------------------------------------------------------
__global__ void __launch_bounds__(256) k_swz_B(const float* __restrict__ W,
                                               _Float16* __restrict__ out) {
  int tid  = blockIdx.x * blockDim.x + threadIdx.x;
  int lane = tid & 31;
  int tile = tid >> 5;                                // tile = kt*256 + nt
  int nt = tile & 255;
  int n  = nt*16 + (lane & 15);
  int kb = (tile >> 8)*32;
  const float* src = W + (size_t)n*1024 + kb;
  _Float16* dst = out + (size_t)tile*512 + lane*16;
#pragma unroll
  for (int i = 0; i < 16; ++i) dst[i] = (_Float16)src[(i < 8) ? ((lane >> 4)*8 + i) : (16 + (lane >> 4)*8 + (i - 8))];
}

// ---------------------------------------------------------------------------
// Input-projection GEMM: G[dir] = X @ W_ih[dir]^T + b[dir], f32 acc -> f16.
// 8 waves/block; each wave: 2 M-tiles x 4 N-tiles (8 WMMA / 6 tile-loads per
// k-step), software-pipelined one k-tile ahead, pinned with sched_barrier.
// ---------------------------------------------------------------------------
__global__ void __launch_bounds__(256) k_gemm_gates(
    const _Float16* __restrict__ Ax, const _Float16* __restrict__ Bih,
    const float* __restrict__ bf, const float* __restrict__ bb,
    _Float16* __restrict__ G) {
  int wave = threadIdx.x >> 5, lane = threadIdx.x & 31;
  int dir  = blockIdx.z;
  int mt0  = (blockIdx.x*8 + wave)*2;      // m-tile pair: mt0, mt0+1
  int nt0  = blockIdx.y*4;                 // 0..252
  const _Float16* Bd   = Bih + (size_t)dir * 4194304;
  const float*    bias = dir ? bb : bf;

  const _Float16* A0 = Ax + (size_t)mt0*32*512 + lane*16;
  const _Float16* A1 = A0 + (size_t)32*512;
  const _Float16* Bb = Bd + (size_t)nt0*512 + lane*16;

  v8f c00 = {}, c01 = {}, c02 = {}, c03 = {};
  v8f c10 = {}, c11 = {}, c12 = {}, c13 = {};
  // prologue: k-tile 0 in registers
  v16h a0 = *(const v16h*)(A0);
  v16h a1 = *(const v16h*)(A1);
  v16h b0 = *(const v16h*)(Bb + 0*512);
  v16h b1 = *(const v16h*)(Bb + 1*512);
  v16h b2 = *(const v16h*)(Bb + 2*512);
  v16h b3 = *(const v16h*)(Bb + 3*512);
  for (int kt = 0; kt < 32; ++kt) {
    v16h a0n = a0, a1n = a1, b0n = b0, b1n = b1, b2n = b2, b3n = b3;
    if (kt < 31) {                                  // issue next k-tile's loads
      size_t ko = (size_t)(kt + 1);
      a0n = *(const v16h*)(A0 + ko*512);
      a1n = *(const v16h*)(A1 + ko*512);
      const _Float16* bk = Bb + ko*131072;          // 256 tiles * 512 halves
      b0n = *(const v16h*)(bk + 0*512);
      b1n = *(const v16h*)(bk + 1*512);
      b2n = *(const v16h*)(bk + 2*512);
      b3n = *(const v16h*)(bk + 3*512);
      __builtin_prefetch(A0 + (ko + 1)*512, 0, 1);
    }
    SCHED_FENCE();                                  // keep loads above the WMMA chain
    c00 = WMMA_F16(a0, b0, c00);
    c01 = WMMA_F16(a0, b1, c01);
    c02 = WMMA_F16(a0, b2, c02);
    c03 = WMMA_F16(a0, b3, c03);
    c10 = WMMA_F16(a1, b0, c10);
    c11 = WMMA_F16(a1, b1, c11);
    c12 = WMMA_F16(a1, b2, c12);
    c13 = WMMA_F16(a1, b3, c13);
    a0 = a0n; a1 = a1n; b0 = b0n; b1 = b1n; b2 = b2n; b3 = b3n;
  }
  // C/D layout: lanes 0-15 hold m=v, lanes 16-31 hold m=v+8; n = lane%16.
  int m0 = (lane >> 4)*8;
  int n  = lane & 15;
  int c0 = nt0*16 + n;
#pragma unroll
  for (int v = 0; v < 8; ++v) {
    int row0 = mt0*16 + m0 + v;
    size_t base0 = ((size_t)dir*ROWS + row0) * (size_t)G4;
    G[base0 + c0     ] = (_Float16)(c00[v] + bias[c0     ]);
    G[base0 + c0 + 16] = (_Float16)(c01[v] + bias[c0 + 16]);
    G[base0 + c0 + 32] = (_Float16)(c02[v] + bias[c0 + 32]);
    G[base0 + c0 + 48] = (_Float16)(c03[v] + bias[c0 + 48]);
    int row1 = (mt0 + 1)*16 + m0 + v;
    size_t base1 = ((size_t)dir*ROWS + row1) * (size_t)G4;
    G[base1 + c0     ] = (_Float16)(c10[v] + bias[c0     ]);
    G[base1 + c0 + 16] = (_Float16)(c11[v] + bias[c0 + 16]);
    G[base1 + c0 + 32] = (_Float16)(c12[v] + bias[c0 + 32]);
    G[base1 + c0 + 48] = (_Float16)(c13[v] + bias[c0 + 48]);
  }
}

// ---------------------------------------------------------------------------
// Persistent BiLSTM recurrence. 128 WGs x 1 wave. WG (dir, w) owns hidden
// units [w*16, w*16+16). W_hh slice (128KB) resident in LDS, staged via TDM.
// h state: double-buffered A-swizzled global buffer; per-step grid barrier.
// ---------------------------------------------------------------------------
__global__ void __launch_bounds__(32) k_bilstm_rec(
    const _Float16* __restrict__ G, const _Float16* __restrict__ Bhh,
    const int* __restrict__ lengths,
    _Float16* __restrict__ hswz, _Float16* __restrict__ hs,
    unsigned* __restrict__ ctr) {
  extern __shared__ _Float16 lds[];                 // 4*32*512 halves = 128 KB
  const int lane = threadIdx.x;
  const int dir  = blockIdx.x >> 6;
  const int w    = blockIdx.x & 63;
  const int j0   = w * 16;

  // Stage resident W_hh slice: nt in {w, 64+w, 128+w, 192+w}, kt = 0..31.
  const _Float16* Bd = Bhh + (size_t)dir * 4194304;
#if __has_builtin(__builtin_amdgcn_tensor_load_to_lds)
#pragma unroll
  for (int g = 0; g < 4; ++g)
    tdm_load_whh_slab((const char*)Bd + (size_t)(g*64 + w)*1024, (unsigned)(g*32768));
#if __has_builtin(__builtin_amdgcn_s_wait_tensorcnt)
  __builtin_amdgcn_s_wait_tensorcnt(0);
#else
  asm volatile("s_wait_tensorcnt 0x0" ::: "memory");
#endif
#else
#pragma unroll
  for (int g = 0; g < 4; ++g) {
    int nt = g*64 + w;
    for (int kt = 0; kt < 32; ++kt) {
      v16h t = *(const v16h*)(Bd + ((size_t)(kt*256 + nt))*512 + lane*16);
      *(v16h*)(lds + (size_t)(g*32 + kt)*512 + lane*16) = t;
    }
  }
#endif
  __syncthreads();

  int lens[8];
#pragma unroll
  for (int m = 0; m < 8; ++m) lens[m] = lengths[m];
  float cc[8], hh[8];
#pragma unroll
  for (int m = 0; m < 8; ++m) { cc[m] = 0.0f; hh[m] = 0.0f; }

  _Float16* hs_d = hs + (size_t)dir * 4194304;      // [8][512][1024]
  const _Float16* Gd = G + (size_t)dir * ROWS * (size_t)G4;
  const unsigned nwg = gridDim.x;

  int   pm[8];
  float gxi[8], gxf[8], gxg[8], gxo[8];

  for (int s = 0; s < LT; ++s) {
    const _Float16* rb = hswz + (size_t)(s & 1)*32768 + dir*16384;        // read buf
    _Float16*       wb = hswz + (size_t)((s + 1) & 1)*32768 + dir*16384;  // write buf

    // Hoist gate-input loads above the WMMA loop: independent of matrix work,
    // so they overlap the 128-WMMA chain instead of stalling after it.
    if (lane < 16) {
      const int j = j0 + lane;
#pragma unroll
      for (int m = 0; m < 8; ++m) {
        const int len = lens[m];
        const bool mk = s < len;
        pm[m] = (dir == 0) ? s : (mk ? (len - 1 - s) : s);
        const size_t gb = ((size_t)(m*LT + pm[m])) * (size_t)G4;
        gxi[m] = (float)Gd[gb + j];
        gxf[m] = (float)Gd[gb + HD   + j];
        gxg[m] = (float)Gd[gb + 2*HD + j];
        gxo[m] = (float)Gd[gb + 3*HD + j];
      }
    }

    // h @ W_hh^T for the 4 gate tiles, h A-tiles pipelined one k-tile ahead.
    v8f a0 = {}, a1 = {}, a2 = {}, a3 = {};
    v16h a_c = *(const v16h*)(rb + lane*16);
    for (int kt = 0; kt < 32; ++kt) {
      v16h a_n = a_c;
      if (kt < 31) a_n = *(const v16h*)(rb + (kt + 1)*512 + lane*16);
      v16h b0 = *(const v16h*)(lds + (size_t)(0*32 + kt)*512 + lane*16);
      v16h b1 = *(const v16h*)(lds + (size_t)(1*32 + kt)*512 + lane*16);
      v16h b2 = *(const v16h*)(lds + (size_t)(2*32 + kt)*512 + lane*16);
      v16h b3 = *(const v16h*)(lds + (size_t)(3*32 + kt)*512 + lane*16);
      SCHED_FENCE();                                // loads stay above WMMA chain
      a0 = WMMA_F16(a_c, b0, a0);
      a1 = WMMA_F16(a_c, b1, a1);
      a2 = WMMA_F16(a_c, b2, a2);
      a3 = WMMA_F16(a_c, b3, a3);
      a_c = a_n;
    }

    if (lane < 16) {                                 // lanes 0-15 hold rows m=0..7
      const int j = j0 + lane;
      const int kl = j & 31, kth = j >> 5;
      const int grp = kl >> 3;
      const int lt  = ((grp & 1) << 4);
      const int idx = (kl & 7) + ((grp >> 1) << 3);
#pragma unroll
      for (int m = 0; m < 8; ++m) {
        const bool mk = s < lens[m];
        float gi = a0[m] + gxi[m];
        float gf = a1[m] + gxf[m];
        float gg = a2[m] + gxg[m];
        float go = a3[m] + gxo[m];
        float cn = sigm(gf)*cc[m] + sigm(gi)*tanhf(gg);
        float hn = sigm(go)*tanhf(cn);
        hs_d[(size_t)(m*LT + pm[m])*HD + j] = (_Float16)(mk ? hn : 0.0f);
        if (mk) { cc[m] = cn; hh[m] = hn; }
        wb[kth*512 + (m + lt)*16 + idx] = (_Float16)hh[m];  // carry state forward
      }
    }

    // grid barrier: all 128 WGs finish step s before anyone reads buf for s+1
    __threadfence();
    if (lane == 0) {
      __hip_atomic_fetch_add(ctr, 1u, __ATOMIC_ACQ_REL, __HIP_MEMORY_SCOPE_AGENT);
      const unsigned target = (unsigned)(s + 1) * nwg;
      while (__hip_atomic_load(ctr, __ATOMIC_ACQUIRE, __HIP_MEMORY_SCOPE_AGENT) < target)
        __builtin_amdgcn_s_sleep(2);
    }
    __syncthreads();
    __threadfence();
  }
}

// ---------------------------------------------------------------------------
// emit[b,t,k] = hf . We[k,:H] + hb . We[k,H:] + be[k]   (K=16, tiny)
// ---------------------------------------------------------------------------
__global__ void __launch_bounds__(256) k_emit(
    const _Float16* __restrict__ hs, const float* __restrict__ We,
    const float* __restrict__ be, float* __restrict__ emit) {
  __shared__ float red[256];
  const int r = blockIdx.x;                         // b*512 + t
  const int k = threadIdx.x & 15;
  const int g = threadIdx.x >> 4;
  const _Float16* hf = hs + (size_t)r * HD;
  const _Float16* hb = hs + (size_t)4194304 + (size_t)r * HD;
  const float* wk = We + (size_t)k * (2*HD);
  float acc = 0.0f;
  for (int j = g; j < HD; j += 16)
    acc += (float)hf[j]*wk[j] + (float)hb[j]*wk[HD + j];
  red[threadIdx.x] = acc;
  __syncthreads();
  for (int s = 8; s > 0; s >>= 1) {
    if (g < s) red[threadIdx.x] += red[threadIdx.x + s*16];
    __syncthreads();
  }
  if (g == 0) emit[(size_t)r*KT_ + k] = red[k] + be[k];
}

// ---------------------------------------------------------------------------
// CRF: gold score + forward-algorithm logZ, out[b] = logZ - total. One wave/b.
// ---------------------------------------------------------------------------
__global__ void __launch_bounds__(32) k_crf(
    const float* __restrict__ emit, const int* __restrict__ tags,
    const int* __restrict__ lengths, const float* __restrict__ trans,
    float* __restrict__ out) {
  __shared__ float T[KT_*KT_];
  __shared__ float d[KT_], dn[KT_];
  __shared__ float tot;
  const int b = blockIdx.x, lane = threadIdx.x;
  for (int i = lane; i < KT_*KT_; i += 32) T[i] = trans[i];
  const int len = lengths[b];
  if (lane == 0) {
    float s = 0.0f; int prev = 0;
    for (int t = 0; t < LT; ++t) {
      if (t < len) {
        int tg = tags[b*LT + t];
        s += emit[((size_t)(b*LT + t))*KT_ + tg];
        if (t > 0) s += T[prev*KT_ + tg];
        prev = tg;
      }
    }
    tot = s;
  }
  if (lane < KT_) d[lane] = emit[((size_t)(b*LT))*KT_ + lane];
  __syncthreads();
  for (int t = 1; t < LT; ++t) {
    if (t < len) {
      if (lane < KT_) {
        float mx = -INFINITY;
        for (int j = 0; j < KT_; ++j) mx = fmaxf(mx, d[j] + T[j*KT_ + lane]);
        float s = 0.0f;
        for (int j = 0; j < KT_; ++j) s += expf(d[j] + T[j*KT_ + lane] - mx);
        dn[lane] = mx + logf(s) + emit[((size_t)(b*LT + t))*KT_ + lane];
      }
      __syncthreads();
      if (lane < KT_) d[lane] = dn[lane];
      __syncthreads();
    }
  }
  if (lane == 0) {
    float mx = -INFINITY;
    for (int k = 0; k < KT_; ++k) mx = fmaxf(mx, d[k]);
    float s = 0.0f;
    for (int k = 0; k < KT_; ++k) s += expf(d[k] - mx);
    out[b] = (mx + logf(s)) - tot;
  }
}

// ---------------------------------------------------------------------------
extern "C" void kernel_launch(void* const* d_in, const int* in_sizes, int n_in,
                              void* d_out, int out_size, void* d_ws, size_t ws_size,
                              hipStream_t stream) {
  const float* x       = (const float*)d_in[0];
  const int*   tags    = (const int*)  d_in[1];
  const int*   lengths = (const int*)  d_in[2];
  const float* W_ih_f  = (const float*)d_in[3];
  const float* W_hh_f  = (const float*)d_in[4];
  const float* b_f     = (const float*)d_in[5];
  const float* W_ih_b  = (const float*)d_in[6];
  const float* W_hh_b  = (const float*)d_in[7];
  const float* b_b     = (const float*)d_in[8];
  const float* W_emit  = (const float*)d_in[9];
  const float* b_emit  = (const float*)d_in[10];
  const float* trans   = (const float*)d_in[11];
  float* out = (float*)d_out;

  char* ws = (char*)d_ws;
  _Float16* Ax   = (_Float16*)(ws + OFF_AX);
  _Float16* Bih  = (_Float16*)(ws + OFF_BIH);
  _Float16* Bhh  = (_Float16*)(ws + OFF_BHH);
  _Float16* G    = (_Float16*)(ws + OFF_G);
  _Float16* hs   = (_Float16*)(ws + OFF_HS);
  _Float16* hswz = (_Float16*)(ws + OFF_HSWZ);
  float*    emit = (float*)   (ws + OFF_EMIT);
  unsigned* ctr  = (unsigned*)(ws + OFF_CTR);

  // zero h-state double buffer + barrier counter (graph-capture safe)
  hipMemsetAsync(ws + OFF_HSWZ, 0, SZ_HSWZ, stream);
  hipMemsetAsync(ws + OFF_CTR,  0, SZ_CTR,  stream);

  // prep swizzles
  k_swz_A<<<1024, 256, 0, stream>>>(x, Ax);
  k_swz_B<<<1024, 256, 0, stream>>>(W_ih_f, Bih);
  k_swz_B<<<1024, 256, 0, stream>>>(W_ih_b, Bih + (size_t)4194304);
  k_swz_B<<<1024, 256, 0, stream>>>(W_hh_f, Bhh);
  k_swz_B<<<1024, 256, 0, stream>>>(W_hh_b, Bhh + (size_t)4194304);

  // input-projection GEMMs (both directions): 16 x 64 x 2 blocks, 8 waves each
  k_gemm_gates<<<dim3(16, 64, 2), 256, 0, stream>>>(Ax, Bih, b_f, b_b, G);

  // persistent recurrence: 128 WGs, 128 KB dynamic LDS each
  k_bilstm_rec<<<128, 32, 4*32*512*sizeof(_Float16), stream>>>(G, Bhh, lengths, hswz, hs, ctr);

  // emission scores + CRF
  k_emit<<<ROWS, 256, 0, stream>>>(hs, W_emit, b_emit, emit);
  k_crf<<<NB, 32, 0, stream>>>(emit, tags, lengths, trans, out);

  (void)in_sizes; (void)n_in; (void)out_size; (void)ws_size;
}
